// SelfAttention_79130477462036
// MI455X (gfx1250) — compile-verified, matrix operand order checked by reading, not run on hardware
//
#include <hip/hip_runtime.h>
#include <hip/hip_bf16.h>

// ---------------------------------------------------------------------------
// Self-attention block (GroupNorm -> dwconv3 -> QKV -> MHSA -> out-proj + res)
// for MI455X / gfx1250. All contractions on v_wmma_f32_16x16x32_bf16.
// Intermediates stored channel-minor ([b][l][c]) so LDS staging is contiguous
// 16B transactions, eligible for GLOBAL_LOAD_ASYNC_TO_LDS_B128.
// B=8, C=256, L=1024, H=8, D=32, GROUPS=32.
// ---------------------------------------------------------------------------

typedef __attribute__((ext_vector_type(4)))  float          v4f;
typedef __attribute__((ext_vector_type(8)))  float          v8f;
typedef __attribute__((ext_vector_type(8)))  unsigned short v8us;
typedef __attribute__((ext_vector_type(16))) unsigned short v16us;
typedef __attribute__((ext_vector_type(16))) __bf16         v16bf;
typedef int gv4i __attribute__((vector_size(16)));  // matches builtin param

#define BATCH 8
#define CCH   256
#define LEN   1024
#define NHEAD 8
#define HDIM  32
#define NGRP  32

#if defined(__gfx1250__) &&                                        \
    __has_builtin(__builtin_amdgcn_global_load_async_to_lds_b128) && \
    __has_builtin(__builtin_amdgcn_s_wait_asynccnt)
#define USE_ASYNC_COPY 1
#else
#define USE_ASYNC_COPY 0
#endif

__device__ __forceinline__ unsigned short f32_to_bf16(float f) {
  unsigned int u = __builtin_bit_cast(unsigned int, f);
  u += 0x7FFFu + ((u >> 16) & 1u);  // round-to-nearest-even
  return (unsigned short)(u >> 16);
}

// Stage 16 contiguous bytes global -> LDS (async if available).
__device__ __forceinline__ void stage16(const unsigned short* g,
                                        unsigned short* l) {
#if USE_ASYNC_COPY
  __builtin_amdgcn_global_load_async_to_lds_b128(
      (__attribute__((address_space(1))) gv4i*)(g),
      (__attribute__((address_space(3))) gv4i*)(l), 0, 0);
#else
  *(v8us*)l = *(const v8us*)g;
#endif
}

__device__ __forceinline__ void stage_wait() {
#if USE_ASYNC_COPY
  __builtin_amdgcn_s_wait_asynccnt(0);
#endif
}

// Build A/B fragment from a row-major LDS tile (per-lane row pre-selected).
// Documented CDNA5 striping (ISA 7.12.2):
//   elements 0..7  -> k = kOff + laneHi*8 + 0..7
//   elements 8..15 -> k = kOff + 16 + laneHi*8 + 0..7
__device__ __forceinline__ v16bf make_frag(const unsigned short* rowBase,
                                           int laneHi, int kOff) {
  v8us lo = *(const v8us*)(rowBase + kOff + laneHi * 8);
  v8us hi = *(const v8us*)(rowBase + kOff + 16 + laneHi * 8);
  union { v16us v; v8us h[2]; } u;
  u.h[0] = lo;
  u.h[1] = hi;
  return __builtin_bit_cast(v16bf, u.v);
}

// ---------------------------------------------------------------------------
// Kernel 1: GroupNorm (32 groups) + depthwise conv1d(k=3, pad=1).
// One block per (batch, group); 8x1024 tile in LDS. Output bf16, layout
// HbfT[b][l][c] (channel-minor) so downstream staging is contiguous.
// ---------------------------------------------------------------------------
__global__ __launch_bounds__(256) void k_gn_dwconv(
    const float* __restrict__ x, const float* __restrict__ gamma,
    const float* __restrict__ beta, const float* __restrict__ dw_w,
    unsigned short* __restrict__ HbfT) {
  __shared__ alignas(16) float sh[8 * LEN];
  __shared__ float red[32];

  const int blk = blockIdx.x;
  const int b = blk >> 5;
  const int g = blk & 31;
  const float* xb = x + ((size_t)(b * CCH + g * 8)) * LEN;

  float s = 0.f, ss = 0.f;
#pragma unroll
  for (int it = 0; it < 32; ++it) {
    int i = threadIdx.x + it * 256;
    float v = xb[i];
    sh[i] = v;
    s += v;
    ss += v * v;
  }
#pragma unroll
  for (int off = 16; off > 0; off >>= 1) {
    s  += __shfl_down(s, off, 32);
    ss += __shfl_down(ss, off, 32);
  }
  const int wave = threadIdx.x >> 5;
  const int lane = threadIdx.x & 31;
  if (lane == 0) { red[wave] = s; red[8 + wave] = ss; }
  __syncthreads();
  if (threadIdx.x == 0) {
    float ts = 0.f, tss = 0.f;
#pragma unroll
    for (int w = 0; w < 8; ++w) { ts += red[w]; tss += red[8 + w]; }
    const float inv_n = 1.0f / (8.0f * LEN);
    float mu = ts * inv_n;
    float var = tss * inv_n - mu * mu;
    red[16] = mu;
    red[17] = rsqrtf(var + 1e-5f);
  }
  __syncthreads();
  const float mu = red[16], rstd = red[17];

#pragma unroll
  for (int it = 0; it < 32; ++it) {
    int i = threadIdx.x + it * 256;
    int c = i >> 10;
    int gc = g * 8 + c;
    sh[i] = (sh[i] - mu) * rstd * gamma[gc] + beta[gc];
  }
  __syncthreads();

#pragma unroll
  for (int it = 0; it < 32; ++it) {
    int i = threadIdx.x + it * 256;
    int c = i >> 10;
    int l = i & (LEN - 1);
    int gc = g * 8 + c;
    float w0 = dw_w[gc * 3 + 0];
    float w1 = dw_w[gc * 3 + 1];
    float w2 = dw_w[gc * 3 + 2];
    float a = (l > 0) ? sh[c * LEN + l - 1] : 0.f;
    float m = sh[i];
    float z = (l < LEN - 1) ? sh[c * LEN + l + 1] : 0.f;
    float y = w0 * a + w1 * m + w2 * z;
    HbfT[((size_t)b * LEN + l) * CCH + gc] = f32_to_bf16(y);
  }
}

// ---------------------------------------------------------------------------
// Kernel 2: QKV projection. GEMM per batch: M=768(m), K=256(c), N=1024(l).
// 64x64 tiles, 8 waves, 2 accumulators per wave.
// Output channel-minor: QKVT[b][l][m].
// ---------------------------------------------------------------------------
__global__ __launch_bounds__(256) void k_gemm_qkv(
    const float* __restrict__ W, const float* __restrict__ bias,
    const unsigned short* __restrict__ HbfT, unsigned short* __restrict__ QKVT) {
  __shared__ alignas(16) unsigned short Wt[64][32];  // [m][k]
  __shared__ alignas(16) unsigned short Ht[64][32];  // [n=l][k=c]

  const int n0 = blockIdx.x * 64;
  const int m0 = blockIdx.y * 64;
  const int b = blockIdx.z;
  const int tid = threadIdx.x;
  const int lane = tid & 31, wavei = tid >> 5;
  const int laneHi = lane >> 4, lane15 = lane & 15;
  const int mt = (wavei & 3) * 16;
  const int ntB = (wavei >> 2) * 32;
  const int sr = tid >> 2;              // staging row 0..63
  const int sc8 = (tid & 3) * 8;        // staging col-octet

  v8f acc0 = {};
  v8f acc1 = {};

  for (int kk = 0; kk < CCH; kk += 32) {
    if (kk + 32 < CCH)
      __builtin_prefetch(
          &HbfT[((size_t)b * LEN + n0 + sr) * CCH + kk + 32 + sc8], 0, 1);
    // A tile: fp32 weights -> bf16, one b128 store per thread.
    {
      const float* src = W + (size_t)(m0 + sr) * CCH + kk + sc8;
      v4f lo = *(const v4f*)src;
      v4f hi = *(const v4f*)(src + 4);
      v8us p;
#pragma unroll
      for (int j = 0; j < 4; ++j) {
        p[j] = f32_to_bf16(lo[j]);
        p[j + 4] = f32_to_bf16(hi[j]);
      }
      *(v8us*)&Wt[sr][sc8] = p;
    }
    // B tile: contiguous 16B copy (async if available).
    stage16(&HbfT[((size_t)b * LEN + n0 + sr) * CCH + kk + sc8], &Ht[sr][sc8]);
    stage_wait();
    __syncthreads();

    v16bf af = make_frag(&Wt[mt + lane15][0], laneHi, 0);
    v16bf b0 = make_frag(&Ht[ntB + lane15][0], laneHi, 0);
    v16bf b1 = make_frag(&Ht[ntB + 16 + lane15][0], laneHi, 0);
    acc0 = __builtin_amdgcn_wmma_f32_16x16x32_bf16(false, af, false, b0,
                                                   (short)0, acc0, false, false);
    acc1 = __builtin_amdgcn_wmma_f32_16x16x32_bf16(false, af, false, b1,
                                                   (short)0, acc1, false, false);
    __syncthreads();
  }

#pragma unroll
  for (int i = 0; i < 8; ++i) {
    int m = m0 + mt + i + laneHi * 8;
    int l = n0 + ntB + lane15;
    float bv = bias[m];
    QKVT[((size_t)b * LEN + l) * (3 * CCH) + m]      = f32_to_bf16(acc0[i] + bv);
    QKVT[((size_t)b * LEN + l + 16) * (3 * CCH) + m] = f32_to_bf16(acc1[i] + bv);
  }
}

// ---------------------------------------------------------------------------
// Kernel 3: flash-style attention per (b, head, 64 query rows).
// S = (Q^T K)/16 (single WMMA k-step, K=D=32); online softmax over 16 key
// tiles; O += P V^T (two k-steps). QKVT is [b][l][m] so Q/K tiles stage as
// contiguous 16B copies; V tile is the one transposed staging.
// ---------------------------------------------------------------------------
__global__ __launch_bounds__(256) void k_attention(
    const unsigned short* __restrict__ QKVT, unsigned short* __restrict__ ObfT) {
  __shared__ alignas(16) unsigned short Qs[64][32];   // [l][d]   A for S
  __shared__ alignas(16) unsigned short KsT[64][32];  // [L'][d]  B for S
  __shared__ alignas(16) unsigned short Vsd[32][64];  // [d][L']  B for PV
  __shared__ alignas(16) float Ss[64][64];
  __shared__ alignas(16) unsigned short Ps[64][64];
  __shared__ float mrow[64], lrow[64], arow[64];

  const int l0 = blockIdx.x * 64;
  const int h = blockIdx.y;
  const int b = blockIdx.z;
  const int tid = threadIdx.x;
  const int lane = tid & 31, wavei = tid >> 5;
  const int laneHi = lane >> 4, lane15 = lane & 15;

  const unsigned short* qkvb = QKVT + (size_t)b * LEN * (3 * CCH);
  const int hq = h * 3 * HDIM;

  const int sr = tid >> 2;            // 0..63
  const int sd8 = (tid & 3) * 8;      // 0,8,16,24
  const int vn = tid & 63;            // V staging column
  const int vd8 = (tid >> 6) * 8;     // V staging d-octet

  // Q^T tile: contiguous copy.
  stage16(&qkvb[(size_t)(l0 + sr) * (3 * CCH) + hq + sd8], &Qs[sr][sd8]);
  if (tid < 64) { mrow[tid] = -1e30f; lrow[tid] = 0.f; }

  const int sm = (wavei & 3) * 16;
  const int snB = (wavei >> 2) * 32;
  const int om = (wavei & 3) * 16;
  const int on = (wavei >> 2) * 16;
  v8f oacc = {};

  for (int j = 0; j < LEN / 64; ++j) {
    const int kbase = j * 64;
    // K tile: contiguous copy.
    stage16(&qkvb[(size_t)(kbase + sr) * (3 * CCH) + hq + HDIM + sd8],
            &KsT[sr][sd8]);
    // V tile: b128 global load + 8-way LDS transpose scatter.
    {
      v8us tv = *(const v8us*)&qkvb[(size_t)(kbase + vn) * (3 * CCH) + hq +
                                    2 * HDIM + vd8];
#pragma unroll
      for (int t = 0; t < 8; ++t) Vsd[vd8 + t][vn] = tv[t];
    }
    stage_wait();
    __syncthreads();

    // ---- S = Q^T K, scaled by 1/sqrt(C)=1/16, 2 subtiles per wave ----
    {
      v16bf aq = make_frag(&Qs[sm + lane15][0], laneHi, 0);
      v16bf k0 = make_frag(&KsT[snB + lane15][0], laneHi, 0);
      v16bf k1 = make_frag(&KsT[snB + 16 + lane15][0], laneHi, 0);
      v8f s0 = {};
      v8f s1 = {};
      s0 = __builtin_amdgcn_wmma_f32_16x16x32_bf16(false, aq, false, k0,
                                                   (short)0, s0, false, false);
      s1 = __builtin_amdgcn_wmma_f32_16x16x32_bf16(false, aq, false, k1,
                                                   (short)0, s1, false, false);
#pragma unroll
      for (int i = 0; i < 8; ++i) {
        int r = sm + i + laneHi * 8;
        Ss[r][snB + lane15]      = s0[i] * 0.0625f;
        Ss[r][snB + 16 + lane15] = s1[i] * 0.0625f;
      }
    }
    __syncthreads();

    // ---- online softmax: 4 lanes per row, wave32 shfl_xor combines ----
    {
      const int r = tid >> 2;
      const int q0 = (tid & 3) * 16;
      float mx = -1e30f;
#pragma unroll
      for (int c = 0; c < 16; ++c) mx = fmaxf(mx, Ss[r][q0 + c]);
      mx = fmaxf(mx, __shfl_xor(mx, 1, 32));
      mx = fmaxf(mx, __shfl_xor(mx, 2, 32));
      float mold = mrow[r];
      mx = fmaxf(mx, mold);
      float alpha = __expf(mold - mx);
      float sum = 0.f;
#pragma unroll
      for (int c = 0; c < 16; ++c) {
        float p = __expf(Ss[r][q0 + c] - mx);
        Ps[r][q0 + c] = f32_to_bf16(p);
        sum += p;
      }
      sum += __shfl_xor(sum, 1, 32);
      sum += __shfl_xor(sum, 2, 32);
      if ((tid & 3) == 0) {
        lrow[r] = lrow[r] * alpha + sum;
        mrow[r] = mx;
        arow[r] = alpha;
      }
    }
    __syncthreads();

    // ---- rescale running O, then O += P V^T (K=64 -> 2 k-steps) ----
#pragma unroll
    for (int i = 0; i < 8; ++i) oacc[i] *= arow[om + i + laneHi * 8];
#pragma unroll
    for (int ks = 0; ks < 2; ++ks) {
      v16bf ap = make_frag(&Ps[om + lane15][0], laneHi, ks * 32);
      v16bf bv = make_frag(&Vsd[on + lane15][0], laneHi, ks * 32);
      oacc = __builtin_amdgcn_wmma_f32_16x16x32_bf16(false, ap, false, bv,
                                                     (short)0, oacc, false, false);
    }
    __syncthreads();
  }

#pragma unroll
  for (int i = 0; i < 8; ++i) {
    int r = om + i + laneHi * 8;
    int d = on + lane15;
    float v = oacc[i] / lrow[r];
    ObfT[((size_t)b * LEN + l0 + r) * CCH + h * HDIM + d] = f32_to_bf16(v);
  }
}

// ---------------------------------------------------------------------------
// Kernel 4: out projection + bias + residual. M=K=256, N=1024, fp32 output
// in reference layout [B, C, L].
// ---------------------------------------------------------------------------
__global__ __launch_bounds__(256) void k_gemm_out(
    const float* __restrict__ W, const float* __restrict__ bias,
    const unsigned short* __restrict__ ObfT, const float* __restrict__ X,
    float* __restrict__ Out) {
  __shared__ alignas(16) unsigned short Wt[64][32];
  __shared__ alignas(16) unsigned short Ot[64][32];  // [n=l][k=c]

  const int n0 = blockIdx.x * 64;
  const int m0 = blockIdx.y * 64;
  const int b = blockIdx.z;
  const int tid = threadIdx.x;
  const int lane = tid & 31, wavei = tid >> 5;
  const int laneHi = lane >> 4, lane15 = lane & 15;
  const int mt = (wavei & 3) * 16;
  const int ntB = (wavei >> 2) * 32;
  const int sr = tid >> 2;
  const int sc8 = (tid & 3) * 8;

  v8f acc0 = {};
  v8f acc1 = {};

  for (int kk = 0; kk < CCH; kk += 32) {
    if (kk + 32 < CCH)
      __builtin_prefetch(
          &ObfT[((size_t)b * LEN + n0 + sr) * CCH + kk + 32 + sc8], 0, 1);
    {
      const float* src = W + (size_t)(m0 + sr) * CCH + kk + sc8;
      v4f lo = *(const v4f*)src;
      v4f hi = *(const v4f*)(src + 4);
      v8us p;
#pragma unroll
      for (int j = 0; j < 4; ++j) {
        p[j] = f32_to_bf16(lo[j]);
        p[j + 4] = f32_to_bf16(hi[j]);
      }
      *(v8us*)&Wt[sr][sc8] = p;
    }
    stage16(&ObfT[((size_t)b * LEN + n0 + sr) * CCH + kk + sc8], &Ot[sr][sc8]);
    stage_wait();
    __syncthreads();

    v16bf af = make_frag(&Wt[mt + lane15][0], laneHi, 0);
    v16bf b0 = make_frag(&Ot[ntB + lane15][0], laneHi, 0);
    v16bf b1 = make_frag(&Ot[ntB + 16 + lane15][0], laneHi, 0);
    acc0 = __builtin_amdgcn_wmma_f32_16x16x32_bf16(false, af, false, b0,
                                                   (short)0, acc0, false, false);
    acc1 = __builtin_amdgcn_wmma_f32_16x16x32_bf16(false, af, false, b1,
                                                   (short)0, acc1, false, false);
    __syncthreads();
  }

#pragma unroll
  for (int i = 0; i < 8; ++i) {
    int m = m0 + mt + i + laneHi * 8;
    int l = n0 + ntB + lane15;
    size_t base = ((size_t)(b * CCH + m)) * LEN + l;
    Out[base]      = acc0[i] + bias[m] + X[base];
    Out[base + 16] = acc1[i] + bias[m] + X[base + 16];
  }
}

// ---------------------------------------------------------------------------
extern "C" void kernel_launch(void* const* d_in, const int* in_sizes, int n_in,
                              void* d_out, int out_size, void* d_ws,
                              size_t ws_size, hipStream_t stream) {
  const float* x     = (const float*)d_in[0];
  const float* gamma = (const float*)d_in[1];
  const float* beta  = (const float*)d_in[2];
  const float* dw_w  = (const float*)d_in[3];
  const float* qkv_w = (const float*)d_in[4];
  const float* qkv_b = (const float*)d_in[5];
  const float* out_w = (const float*)d_in[6];
  const float* out_b = (const float*)d_in[7];
  float* out = (float*)d_out;

  char* ws = (char*)d_ws;
  const size_t hBytes   = (size_t)BATCH * CCH * LEN * sizeof(unsigned short);
  const size_t qkvBytes = (size_t)BATCH * 3 * CCH * LEN * sizeof(unsigned short);
  unsigned short* HbfT = (unsigned short*)ws;
  unsigned short* QKVT = (unsigned short*)(ws + hBytes);
  unsigned short* ObfT = (unsigned short*)(ws + hBytes + qkvBytes);

  k_gn_dwconv<<<BATCH * NGRP, 256, 0, stream>>>(x, gamma, beta, dw_w, HbfT);
  k_gemm_qkv<<<dim3(LEN / 64, (3 * CCH) / 64, BATCH), 256, 0, stream>>>(
      qkv_w, qkv_b, HbfT, QKVT);
  k_attention<<<dim3(LEN / 64, NHEAD, BATCH), 256, 0, stream>>>(QKVT, ObfT);
  k_gemm_out<<<dim3(LEN / 64, CCH / 64, BATCH), 256, 0, stream>>>(
      out_w, out_b, ObfT, x, out);
}
